// LinearAttention_77773267796307
// MI455X (gfx1250) — matrix operand order repaired and checked by reading, not compiled, not run on hardware
//
#include <hip/hip_runtime.h>
#include <math.h>

#define B_   4
#define S_   4096
#define D_   1024
#define H_   16
#define DH_  64
#define C_   64            // chunk length
#define NC_  (S_ / C_)     // 64 chunks
#define EPS_ 1e-6f

typedef __bf16 bf16;
typedef __attribute__((ext_vector_type(2)))  __bf16 v2bf;
typedef __attribute__((ext_vector_type(8)))  __bf16 v8bf;
typedef __attribute__((ext_vector_type(16))) __bf16 v16bf;
typedef __attribute__((ext_vector_type(8)))  float  v8f;

__device__ __forceinline__ v8f zero_v8f() {
  v8f v;
#pragma unroll
  for (int i = 0; i < 8; ++i) v[i] = 0.0f;
  return v;
}

// ---- CDNA5 async global->LDS copy (ASYNCcnt path) --------------------------
// Generic shared pointers keep the wave-relative LDS offset in ADDR[31:0]
// (flat aperture check uses ADDR[63:32]), so truncation yields the LDS addr.
__device__ __forceinline__ void async_b128(const void* lds_ptr, const void* gptr) {
  unsigned lds_off = (unsigned)(unsigned long long)lds_ptr;
  asm volatile("global_load_async_to_lds_b128 %0, %1, off"
               :: "v"(lds_off), "v"(gptr) : "memory");
}
__device__ __forceinline__ void wait_async0() {
  asm volatile("s_wait_asynccnt 0x0" ::: "memory");
}

// A operand (16x32 bf16), row-major source [m][k], stride mult of 8.
__device__ __forceinline__ v16bf frag_a(const bf16* base, int stride, int lane) {
  const int row = lane & 15;
  const int kb  = (lane >> 4) << 3;
  const bf16* p = base + row * stride + kb;
  v8bf lo = *(const v8bf*)(p);
  v8bf hi = *(const v8bf*)(p + 16);
  return __builtin_shufflevector(lo, hi, 0,1,2,3,4,5,6,7,8,9,10,11,12,13,14,15);
}
// B operand (32x16 bf16), n-major source [n][k].
__device__ __forceinline__ v16bf frag_b(const bf16* base, int stride, int lane) {
  const int n  = lane & 15;
  const int kb = (lane >> 4) << 3;
  const bf16* p = base + n * stride + kb;
  v8bf lo = *(const v8bf*)(p);
  v8bf hi = *(const v8bf*)(p + 16);
  return __builtin_shufflevector(lo, hi, 0,1,2,3,4,5,6,7,8,9,10,11,12,13,14,15);
}
__device__ __forceinline__ v8f wmma_bf16(v16bf a, v16bf b, v8f c) {
  return __builtin_amdgcn_wmma_f32_16x16x32_bf16(false, a, false, b, (short)0, c,
                                                 false, false);
}
__device__ __forceinline__ float phi_elu1(float x) {
  return x > 0.0f ? x + 1.0f : __expf(x);   // elu(x)+1
}
__device__ __forceinline__ v8bf pack8(const float4& a, const float4& b) {
  v8bf v;
  v[0] = (bf16)a.x; v[1] = (bf16)a.y; v[2] = (bf16)a.z; v[3] = (bf16)a.w;
  v[4] = (bf16)b.x; v[5] = (bf16)b.y; v[6] = (bf16)b.z; v[7] = (bf16)b.w;
  return v;
}

// ---------------------------------------------------------------------------
// Pre-pass A: fp32 -> bf16 flat convert (x). 8 elems/thread.
// ---------------------------------------------------------------------------
__global__ __launch_bounds__(256) void k_cvt(const float* __restrict__ src,
                                             bf16* __restrict__ dst) {
  const size_t base = ((size_t)blockIdx.x * 256 + threadIdx.x) * 8;
  float4 a = *(const float4*)(src + base);
  float4 b = *(const float4*)(src + base + 4);
  *(v8bf*)(dst + base) = pack8(a, b);
}

// ---------------------------------------------------------------------------
// Pre-pass B: W (k,n) fp32 -> Wt (n,k) bf16, 64x64 LDS tile transpose.
// grid (K/64, N/64), block 256
// ---------------------------------------------------------------------------
__global__ __launch_bounds__(256) void k_wtrans(const float* __restrict__ W,
                                                bf16* __restrict__ Wt) {
  __shared__ float lt[64 * 65];
  const int tid = threadIdx.x;
  const int k0 = blockIdx.x * 64;
  const int n0 = blockIdx.y * 64;

  const int r  = tid >> 2;              // k-local
  const int c0 = (tid & 3) * 16;        // n-local
#pragma unroll
  for (int i = 0; i < 16; i += 4) {
    float4 f = *(const float4*)(W + (size_t)(k0 + r) * D_ + n0 + c0 + i);
    lt[(c0 + i + 0) * 65 + r] = f.x;
    lt[(c0 + i + 1) * 65 + r] = f.y;
    lt[(c0 + i + 2) * 65 + r] = f.z;
    lt[(c0 + i + 3) * 65 + r] = f.w;
  }
  __syncthreads();
  const int n  = tid >> 2;              // n-local
  const int kc = (tid & 3) * 16;        // k-local
  float4 a, b;
  const float* p = lt + n * 65 + kc;
  a.x = p[0]; a.y = p[1]; a.z = p[2]; a.w = p[3];
  b.x = p[4]; b.y = p[5]; b.z = p[6]; b.w = p[7];
  *(v8bf*)(Wt + (size_t)(n0 + n) * D_ + k0 + kc)     = pack8(a, b);
  a.x = p[8];  a.y = p[9];  a.z = p[10]; a.w = p[11];
  b.x = p[12]; b.y = p[13]; b.z = p[14]; b.w = p[15];
  *(v8bf*)(Wt + (size_t)(n0 + n) * D_ + k0 + kc + 8) = pack8(a, b);
}

// ---------------------------------------------------------------------------
// GEMM: Y = f(X @ Wt^T) with X bf16 [M][K], Wt bf16 [n][k] (pre-transposed).
// Block tile 128x128, 8 waves x 8 WMMA tiles, LDS double-buffered, tiles
// fetched with GLOBAL_LOAD_ASYNC_TO_LDS_B128 (ASYNCcnt), 1 barrier / K-step.
// MODE 0/1: phi -> bf16 head layout; 2: id -> bf16 head layout;
// MODE 3: id -> fp32 flat [M][D]
// grid: (D/128, M/128), block 256
// ---------------------------------------------------------------------------
template <int MODE>
__global__ __launch_bounds__(256) void k_gemm(
    const bf16* __restrict__ X, const bf16* __restrict__ Wt,
    bf16* __restrict__ dstb, float* __restrict__ dstf)
{
  __shared__ __align__(16) bf16 lA[2][128 * 40];   // [m][k], stride 40
  __shared__ __align__(16) bf16 lB[2][128 * 40];   // [n][k], stride 40

  const int tid  = threadIdx.x;
  const int lane = tid & 31;
  const int wm   = tid >> 5;
  const int m0 = blockIdx.y * 128;
  const int n0 = blockIdx.x * 128;

  // tile = 128 rows x 32 bf16 = 128 x 4 chunks of 16B; 512 chunks / 256 thr
  auto issue_tile = [&](int kk, int buf) {
#pragma unroll
    for (int i = 0; i < 2; ++i) {
      const int q   = tid + i * 256;    // 0..511
      const int row = q >> 2;
      const int ch  = (q & 3) * 8;      // bf16 elems within row
      async_b128(lA[buf] + row * 40 + ch, X  + (size_t)(m0 + row) * D_ + kk + ch);
      async_b128(lB[buf] + row * 40 + ch, Wt + (size_t)(n0 + row) * D_ + kk + ch);
    }
  };

  v8f acc[8];
#pragma unroll
  for (int j = 0; j < 8; ++j) acc[j] = zero_v8f();

  issue_tile(0, 0);
  wait_async0();
  __syncthreads();

  const int NSTEP = D_ / 32;
  for (int step = 0; step < NSTEP; ++step) {
    const int cur = step & 1;
    const bool more = (step + 1) < NSTEP;
    if (more) issue_tile((step + 1) * 32, cur ^ 1);   // async into other buffer

    v16bf a = frag_a(lA[cur] + wm * 16 * 40, 40, lane);
    v16bf bfr[8];
#pragma unroll
    for (int j = 0; j < 8; ++j)
      bfr[j] = frag_b(lB[cur] + j * 16 * 40, 40, lane);
#pragma unroll
    for (int j = 0; j < 8; ++j)
      acc[j] = wmma_bf16(a, bfr[j], acc[j]);

    if (more) wait_async0();
    __syncthreads();
  }

  const int hi = lane >> 4, nlo = lane & 15;
  if (MODE == 3) {
#pragma unroll
    for (int j = 0; j < 8; ++j) {
      const int n = n0 + j * 16 + nlo;
#pragma unroll
      for (int r = 0; r < 8; ++r) {
        const int m = m0 + wm * 16 + r + 8 * hi;
        dstf[(size_t)m * D_ + n] = acc[j][r];
      }
    }
  } else {
    const int bb = m0 >> 12;               // block never crosses a batch
    const int s0 = (m0 & (S_ - 1)) + wm * 16;
#pragma unroll
    for (int j = 0; j < 8; ++j) {
      const int n  = n0 + j * 16 + nlo;
      const int hh = n >> 6, dh = n & 63;
      bf16* base = dstb + ((size_t)(bb * H_ + hh) * S_) * DH_ + dh;
#pragma unroll
      for (int r = 0; r < 8; ++r) {
        const int ss = s0 + r + 8 * hi;
        float val = acc[j][r];
        if (MODE < 2) val = phi_elu1(val);
        base[(size_t)ss * DH_] = (bf16)val;
      }
    }
  }
}

// ---------------------------------------------------------------------------
// Kernel 2: per-chunk local state: KtV = K^T V (64x64), kz = colsum(K)
// grid: (NC, B*H), block 256
// ---------------------------------------------------------------------------
__global__ __launch_bounds__(256) void k_chunkstate(
    const bf16* __restrict__ kh, const bf16* __restrict__ vh,
    float* __restrict__ localCS)
{
  __shared__ __align__(16) bf16 lKT[64 * 72];   // K^T: [dh][t]
  __shared__ __align__(16) bf16 lVT[64 * 72];   // V^T: [e][t]

  const int tid  = threadIdx.x;
  const int lane = tid & 31;
  const int wave = tid >> 5;
  const int wm = wave >> 1, wn = wave & 1;
  const int c  = blockIdx.x;
  const int bh = blockIdx.y;

  const bf16* kc = kh + ((size_t)bh * S_ + (size_t)c * C_) * DH_;
  const bf16* vc = vh + ((size_t)bh * S_ + (size_t)c * C_) * DH_;
  float* out = localCS + ((size_t)bh * NC_ + c) * (DH_ + 1) * DH_;

  // transposed fills: pair two consecutive t for one dh -> b32 LDS stores
#pragma unroll
  for (int i = 0; i < 8; ++i) {
    const int p  = tid + i * 256;           // 0..2047
    const int dh = p & 63;
    const int t  = (p >> 6) * 2;
    v2bf k2, v2;
    k2[0] = kc[t * DH_ + dh]; k2[1] = kc[(t + 1) * DH_ + dh];
    v2[0] = vc[t * DH_ + dh]; v2[1] = vc[(t + 1) * DH_ + dh];
    *(v2bf*)(lKT + dh * 72 + t) = k2;
    *(v2bf*)(lVT + dh * 72 + t) = v2;
  }
  if (tid < 64) {
    float s = 0.0f;
    for (int t = 0; t < C_; ++t) s += (float)kc[t * DH_ + tid];
    out[DH_ * DH_ + tid] = s;
  }
  __syncthreads();

  v8f acc[2];
  acc[0] = zero_v8f(); acc[1] = zero_v8f();
#pragma unroll
  for (int ks = 0; ks < 2; ++ks) {
    v16bf a  = frag_a(lKT + wm * 16 * 72 + ks * 32, 72, lane);
    v16bf b0 = frag_b(lVT + (wn * 32 +  0) * 72 + ks * 32, 72, lane);
    v16bf b1 = frag_b(lVT + (wn * 32 + 16) * 72 + ks * 32, 72, lane);
    acc[0] = wmma_bf16(a, b0, acc[0]);
    acc[1] = wmma_bf16(a, b1, acc[1]);
  }

  const int hi = lane >> 4, nlo = lane & 15;
#pragma unroll
  for (int j = 0; j < 2; ++j)
#pragma unroll
    for (int r = 0; r < 8; ++r) {
      const int m = wm * 16 + r + 8 * hi;       // dh of k
      const int n = wn * 32 + j * 16 + nlo;     // dh of v
      out[m * DH_ + n] = acc[j][r];
    }
}

// ---------------------------------------------------------------------------
// Kernel 3: exclusive prefix over chunk states, seeded from state_cache.
// grid: (B*H), block 256
// ---------------------------------------------------------------------------
__global__ __launch_bounds__(256) void k_prefix(
    const float* __restrict__ state_cache,
    const float* __restrict__ localCS,
    float* __restrict__ prefixCS)
{
  const int tid = threadIdx.x;
  const int bh  = blockIdx.x;
  const int NELEM = (DH_ + 1) * DH_;            // 4160
  const float* init = state_cache + (size_t)bh * NELEM;

  float run[17];
#pragma unroll
  for (int i = 0; i < 17; ++i) {
    const int e = tid + i * 256;
    run[i] = (e < NELEM) ? init[e] : 0.0f;
  }
  for (int c = 0; c < NC_; ++c) {
    const size_t base = ((size_t)bh * NC_ + c) * NELEM;
#pragma unroll
    for (int i = 0; i < 17; ++i) {
      const int e = tid + i * 256;
      if (e < NELEM) {
        const float l = localCS[base + e];
        prefixCS[base + e] = run[i];
        run[i] += l;
      }
    }
  }
}

// ---------------------------------------------------------------------------
// Kernel 4: per-chunk output.
//   P = causal(Q K^T);  O = Q @ Sprev + P @ V;  den = q.z_prev + rowsum(P)
// grid: (NC, B*H), block 256 (8 waves: 4 row strips x 2 col halves)
// ---------------------------------------------------------------------------
__global__ __launch_bounds__(256) void k_chunkout(
    const bf16* __restrict__ qh, const bf16* __restrict__ kh,
    const bf16* __restrict__ vh, const float* __restrict__ prefixCS,
    bf16* __restrict__ obuf)
{
  __shared__ __align__(16) bf16 lQ[64 * 72];    // [t][dh]  A operand
  __shared__ __align__(16) bf16 lK[64 * 72];    // [t'][dh] B operand (n=t')
  __shared__ __align__(16) bf16 lVT[64 * 72];   // [e][t']  B operand (n=e)
  __shared__ __align__(16) bf16 lSpT[64 * 72];  // [e][d]   B operand (n=e)
  __shared__ __align__(16) bf16 lP[64 * 72];    // [t][t']  A operand
  __shared__ float lz[64];
  __shared__ float lrowp[2][64];
  __shared__ float lden[64];

  const int tid  = threadIdx.x;
  const int lane = tid & 31;
  const int wave = tid >> 5;
  const int it = wave >> 1;
  const int wn = wave & 1;
  const int c  = blockIdx.x;
  const int bh = blockIdx.y;
  const int hi = lane >> 4, nlo = lane & 15;

  const bf16* qc = qh + ((size_t)bh * S_ + (size_t)c * C_) * DH_;
  const bf16* kc = kh + ((size_t)bh * S_ + (size_t)c * C_) * DH_;
  const bf16* vc = vh + ((size_t)bh * S_ + (size_t)c * C_) * DH_;
  const float* sp = prefixCS + ((size_t)bh * NC_ + c) * (DH_ + 1) * DH_;

  // row-major fills (Q, K): async B128 copies (pure byte moves)
#pragma unroll
  for (int i = 0; i < 2; ++i) {
    const int q  = tid + i * 256;               // 0..511
    const int t  = q >> 3;
    const int ch = (q & 7) * 8;                 // bf16 elems in row
    async_b128(lQ + t * 72 + ch, qc + t * DH_ + ch);
    async_b128(lK + t * 72 + ch, kc + t * DH_ + ch);
  }
  // transposed fills (V^T, Sprev^T): paired b32 stores
#pragma unroll
  for (int i = 0; i < 8; ++i) {
    const int p = tid + i * 256;                // 0..2047
    const int e = p & 63;
    const int t = (p >> 6) * 2;
    v2bf a2, s2;
    a2[0] = vc[t * DH_ + e]; a2[1] = vc[(t + 1) * DH_ + e];
    s2[0] = (bf16)sp[t * DH_ + e]; s2[1] = (bf16)sp[(t + 1) * DH_ + e];
    *(v2bf*)(lVT  + e * 72 + t) = a2;
    *(v2bf*)(lSpT + e * 72 + t) = s2;
  }
  if (tid < 64) lz[tid] = sp[DH_ * DH_ + tid];
  wait_async0();
  __syncthreads();

  // Q A-fragments reused by phase A and phase B
  v16bf aq[2];
  aq[0] = frag_a(lQ + it * 16 * 72 +  0, 72, lane);
  aq[1] = frag_a(lQ + it * 16 * 72 + 32, 72, lane);

  // --- Phase A: P = causal(Q K^T) + per-row sums ---
  float rs[8];
#pragma unroll
  for (int r = 0; r < 8; ++r) rs[r] = 0.0f;

#pragma unroll
  for (int jj0 = 0; jj0 < 2; ++jj0) {
    const int jj = wn * 2 + jj0;
    v8f p = zero_v8f();
    if (jj <= it) {
      v16bf b0 = frag_b(lK + jj * 16 * 72 +  0, 72, lane);
      v16bf b1 = frag_b(lK + jj * 16 * 72 + 32, 72, lane);
      p = wmma_bf16(aq[0], b0, p);
      p = wmma_bf16(aq[1], b1, p);
      if (jj == it) {
#pragma unroll
        for (int r = 0; r < 8; ++r)
          if (nlo > r + 8 * hi) p[r] = 0.0f;    // causal: keep t' <= t
      }
#pragma unroll
      for (int r = 0; r < 8; ++r) rs[r] += p[r];
    }
#pragma unroll
    for (int r = 0; r < 8; ++r)
      lP[(it * 16 + r + 8 * hi) * 72 + jj * 16 + nlo] = (bf16)p[r];
  }
#pragma unroll
  for (int r = 0; r < 8; ++r) {
#pragma unroll
    for (int off = 1; off < 16; off <<= 1)
      rs[r] += __shfl_xor(rs[r], off, 32);
  }
  if (nlo == 0) {
#pragma unroll
    for (int r = 0; r < 8; ++r)
      lrowp[wn][it * 16 + r + 8 * hi] = rs[r];
  }
  __syncthreads();

  // --- den = q.z_prev + rowsum(P) ---
  if (tid < 64) {
    float d = 0.0f;
#pragma unroll
    for (int eg = 0; eg < 8; ++eg) {
      v8bf qv = *(const v8bf*)(lQ + tid * 72 + eg * 8);
#pragma unroll
      for (int u = 0; u < 8; ++u) d += (float)qv[u] * lz[eg * 8 + u];
    }
    lden[tid] = d + lrowp[0][tid] + lrowp[1][tid];
  }

  // --- Phase B: O = Q @ Sprev + P @ V ---
  v8f o[2];
  o[0] = zero_v8f(); o[1] = zero_v8f();
#pragma unroll
  for (int ks = 0; ks < 2; ++ks) {
    v16bf ap  = frag_a(lP + it * 16 * 72 + ks * 32, 72, lane);
    v16bf bs0 = frag_b(lSpT + (wn * 2 + 0) * 16 * 72 + ks * 32, 72, lane);
    v16bf bs1 = frag_b(lSpT + (wn * 2 + 1) * 16 * 72 + ks * 32, 72, lane);
    v16bf bv0 = frag_b(lVT  + (wn * 2 + 0) * 16 * 72 + ks * 32, 72, lane);
    v16bf bv1 = frag_b(lVT  + (wn * 2 + 1) * 16 * 72 + ks * 32, 72, lane);
    o[0] = wmma_bf16(aq[ks], bs0, o[0]);
    o[1] = wmma_bf16(aq[ks], bs1, o[1]);
    o[0] = wmma_bf16(ap, bv0, o[0]);
    o[1] = wmma_bf16(ap, bv1, o[1]);
  }
  __syncthreads();

  const int bb = bh >> 4, hh = bh & 15;
#pragma unroll
  for (int j = 0; j < 2; ++j) {
    const int jj = wn * 2 + j;
#pragma unroll
    for (int r = 0; r < 8; ++r) {
      const int tl = it * 16 + r + 8 * hi;
      const int ss = c * C_ + tl;
      const int e  = jj * 16 + nlo;
      obuf[((size_t)bb * S_ + ss) * D_ + hh * DH_ + e] =
          (bf16)(o[j][r] / (lden[tl] + EPS_));
    }
  }
}

// ---------------------------------------------------------------------------
extern "C" void kernel_launch(void* const* d_in, const int* in_sizes, int n_in,
                              void* d_out, int out_size, void* d_ws, size_t ws_size,
                              hipStream_t stream) {
  const float* x  = (const float*)d_in[0];
  const float* sc = (const float*)d_in[1];   // state_cache (B,H,DH+1,DH)
  const float* Wq = (const float*)d_in[2];
  const float* Wk = (const float*)d_in[3];
  const float* Wv = (const float*)d_in[4];
  const float* Wo = (const float*)d_in[5];
  float* out = (float*)d_out;

  const size_t NQ  = (size_t)B_ * H_ * S_ * DH_;               // 16.78M elems
  const size_t NW  = (size_t)D_ * D_;                          // 1.05M elems
  const size_t NCS = (size_t)B_ * H_ * NC_ * (DH_ + 1) * DH_;  // fp32 states

  float* lcs = (float*)d_ws;
  float* pcs = lcs + NCS;
  bf16*  xb  = (bf16*)(pcs + NCS);
  bf16*  wtq = xb  + NQ;
  bf16*  wtk = wtq + NW;
  bf16*  wtv = wtk + NW;
  bf16*  wto = wtv + NW;
  bf16*  qh  = wto + NW;
  bf16*  kh  = qh + NQ;
  bf16*  vh  = kh + NQ;
  bf16*  ob  = vh + NQ;

  dim3 blk(256);
  // pre-pass: bf16 conversions / weight transposes
  k_cvt<<<dim3((unsigned)(NQ / 2048)), blk, 0, stream>>>(x, xb);
  dim3 tgrid(D_ / 64, D_ / 64);
  k_wtrans<<<tgrid, blk, 0, stream>>>(Wq, wtq);
  k_wtrans<<<tgrid, blk, 0, stream>>>(Wk, wtk);
  k_wtrans<<<tgrid, blk, 0, stream>>>(Wv, wtv);
  k_wtrans<<<tgrid, blk, 0, stream>>>(Wo, wto);

  dim3 ggrid(D_ / 128, (B_ * S_) / 128);
  k_gemm<0><<<ggrid, blk, 0, stream>>>(xb, wtq, qh, nullptr);
  k_gemm<1><<<ggrid, blk, 0, stream>>>(xb, wtk, kh, nullptr);
  k_gemm<2><<<ggrid, blk, 0, stream>>>(xb, wtv, vh, nullptr);
  k_chunkstate<<<dim3(NC_, B_ * H_), blk, 0, stream>>>(kh, vh, lcs);
  k_prefix<<<dim3(B_ * H_), blk, 0, stream>>>(sc, lcs, pcs);
  k_chunkout<<<dim3(NC_, B_ * H_), blk, 0, stream>>>(qh, kh, vh, pcs, ob);
  k_gemm<3><<<ggrid, blk, 0, stream>>>(ob, wto, nullptr, out);
}
